// ContrastAwareAttentionBlock_31250182046261
// MI455X (gfx1250) — matrix operand order repaired and checked by reading, not compiled, not run on hardware
//
#include <hip/hip_runtime.h>
#include <hip/hip_bf16.h>
#include <cstdint>

typedef __attribute__((ext_vector_type(16))) _Float16 v16h;
typedef __attribute__((ext_vector_type(8)))  _Float16 v8h;
typedef __attribute__((ext_vector_type(8)))  float    v8f;

// ---------------------------------------------------------------------------
// CDNA5 helpers
// ---------------------------------------------------------------------------
__device__ __forceinline__ v8f wmma_f16(v16h a, v16h b, v8f c) {
  return __builtin_amdgcn_wmma_f32_16x16x32_f16(
      /*neg_a=*/false, a, /*neg_b=*/false, b,
      /*c_mod=*/(short)0, c, /*reuse_a=*/false, /*reuse_b=*/false);
}

// Low 32 bits of a generic pointer to __shared__ = LDS byte address.
__device__ __forceinline__ unsigned lds_addr_of(const void* p) {
  return (unsigned)(uintptr_t)p;
}

// Async copy 16 bytes global -> LDS (ASYNCcnt-tracked, no VGPR landing zone).
__device__ __forceinline__ void async_copy16(unsigned lds_byte, int goff,
                                             const void* gbase) {
  asm volatile("global_load_async_to_lds_b128 %0, %1, %2 offset:0"
               :: "v"(lds_byte), "v"(goff), "s"(gbase)
               : "memory");
}
__device__ __forceinline__ void wait_async() {
  asm volatile("s_wait_asynccnt 0x0" ::: "memory");
}

// A fragment (16x32 f16 MxK). lane m = lane&15, hi = lane>>4:
// halves i -> k = i + 16*hi.
__device__ __forceinline__ v16h make_afrag(const _Float16* pa, int hiL) {
  v8h lo = *(const v8h*)(pa + 16 * hiL);
  v8h hi = *(const v8h*)(pa + 16 * hiL + 8);
  v16h a;
#pragma unroll
  for (int i = 0; i < 8; ++i) { a[i] = lo[i]; a[i + 8] = hi[i]; }
  return a;
}

// ---------------------------------------------------------------------------
// One-shot weight pack: f32 -> f16, conv taps transposed to [layer][tap][o][c].
// Region layout (halves): [0,147456) conv, [147456,155648) w_v|w_proj,
// [155648,197120) w_attn.
// ---------------------------------------------------------------------------
__global__ __launch_bounds__(256) void pack_weights(
    const float* __restrict__ conv_w, const float* __restrict__ w_v,
    const float* __restrict__ w_proj, const float* __restrict__ w_attn,
    _Float16* __restrict__ dst) {
  int idx = blockIdx.x * 256 + threadIdx.x;   // < 197120
  float val;
  if (idx < 147456) {
    int t = idx;
    int c = t & 63; t >>= 6;
    int o = t & 63; t >>= 6;
    int tap = t % 9, layer = t / 9;
    int ti = tap / 3, tj = tap % 3;
    val = conv_w[((((size_t)layer * 64 + o) * 64 + c) * 3 + ti) * 3 + tj];
  } else if (idx < 155648) {
    int i2 = idx - 147456;
    val = (i2 < 4096) ? w_v[i2] : w_proj[i2 - 4096];
  } else {
    val = w_attn[idx - 155648];
  }
  dst[idx] = (_Float16)val;
}

// ---------------------------------------------------------------------------
// NCHW f32 -> BHWC f16
// ---------------------------------------------------------------------------
__global__ __launch_bounds__(256) void nchw_to_bhwc_h(
    const float* __restrict__ in, _Float16* __restrict__ out) {
  size_t i = (size_t)blockIdx.x * 256 + threadIdx.x;
  int w = (int)(i & 127); size_t t = i >> 7;
  int h = (int)(t & 127); t >>= 7;
  int c = (int)(t & 63);  size_t b = t >> 6;
  out[(((b * 128 + h) * 128 + w) * 64) + c] = (_Float16)in[i];
}

// ---------------------------------------------------------------------------
// 3x3 conv (pad 1) + eval BN + ReLU. f16 BHWC in; out f16 BHWC or f32 NCHW.
// One block per image row. Input rows + packed weights staged via async copy.
// ---------------------------------------------------------------------------
__global__ __launch_bounds__(256) void conv3_bn_relu(
    const _Float16* __restrict__ in,    // BHWC f16
    const _Float16* __restrict__ wpk,   // packed [tap][o][c] f16 (36864 halves)
    const float* __restrict__ g, const float* __restrict__ bb,
    const float* __restrict__ mm, const float* __restrict__ vv,
    void* __restrict__ out, int out_nchw) {
  __shared__ alignas(32) _Float16 lA[3 * 130 * 64];  // 3 rows, padded cols
  __shared__ alignas(32) _Float16 lW[9 * 64 * 64];

  int bh = blockIdx.x;
  int b = bh >> 7, h = bh & 127;
  int tid = threadIdx.x;

  // Interior: async copy each valid row (128*64 f16 = 16KB = 1024 x 16B).
  for (int r = 0; r < 3; ++r) {
    int hh = h + r - 1;
    _Float16* dstRow = &lA[((size_t)r * 130 + 1) * 64];
    if (hh >= 0 && hh < 128) {
      const _Float16* srcRow = in + (((size_t)b * 128 + hh) * 128) * 64;
      unsigned lbase = lds_addr_of(dstRow);
      for (int ck = tid; ck < 1024; ck += 256)
        async_copy16(lbase + ck * 16, ck * 16, srcRow);
    } else {
      v8h z = {};
      for (int ck = tid; ck < 1024; ck += 256)
        *(v8h*)((char*)dstRow + ck * 16) = z;
    }
  }
  // Zero border cols 0 and 129: 3 rows * 128 halves = 48 x v8h.
  {
    v8h z = {};
    for (int idx = tid; idx < 48; idx += 256) {
      int r = idx / 16, s = idx % 16;
      int col = (s < 8) ? 0 : 129;
      *(v8h*)&lA[(((size_t)r * 130 + col) * 64) + (s & 7) * 8] = z;
    }
  }
  // Weights: contiguous async copy, 73728 B = 4608 x 16B.
  {
    unsigned lbase = lds_addr_of(lW);
    for (int ck = tid; ck < 4608; ck += 256)
      async_copy16(lbase + ck * 16, ck * 16, wpk);
  }
  wait_async();
  __syncthreads();

  int wv = tid >> 5, lane = tid & 31;
  int mL = lane & 15, hiL = lane >> 4;
  int m0 = wv * 16;

  v8f acc[4] = {};
  for (int tap = 0; tap < 9; ++tap) {
    int ti = tap / 3, tj = tap % 3;
    const _Float16* arow = &lA[((ti * 130) + (m0 + tj)) * 64];
    const _Float16* wtap = &lW[tap * 4096];
#pragma unroll
    for (int c0 = 0; c0 < 64; c0 += 32) {
      v16h af = make_afrag(arow + (size_t)mL * 64 + c0, hiL);
#pragma unroll
      for (int nt = 0; nt < 4; ++nt) {
        v16h bf = *(const v16h*)&wtap[(size_t)(nt * 16 + mL) * 64 + c0 + 16 * hiL];
        acc[nt] = wmma_f16(af, bf, acc[nt]);
      }
    }
  }

#pragma unroll
  for (int nt = 0; nt < 4; ++nt) {
    int n = nt * 16 + mL;
    float inv = g[n] * rsqrtf(vv[n] + 1e-5f);
    float sh  = bb[n] - mm[n] * inv;
#pragma unroll
    for (int r = 0; r < 8; ++r) {
      int p = m0 + r + 8 * hiL;
      float val = fmaxf(acc[nt][r] * inv + sh, 0.f);
      if (out_nchw)
        ((float*)out)[(((size_t)b * 64 + n) * 128 + h) * 128 + p] = val;
      else
        ((_Float16*)out)[(((size_t)b * 128 + h) * 128 + p) * 64 + n] = (_Float16)val;
    }
  }
}

// ---------------------------------------------------------------------------
// Y[M,64] = X[M,64] @ W^T + bias ; X f16 (async-staged), W packed f16, Y f16.
// ---------------------------------------------------------------------------
__global__ __launch_bounds__(256) void linear64_h(
    const _Float16* __restrict__ in, const _Float16* __restrict__ wpk,
    const float* __restrict__ bias, _Float16* __restrict__ out) {
  __shared__ alignas(32) _Float16 lX[128 * 64];
  __shared__ alignas(32) _Float16 lW[64 * 64];
  size_t m_base = (size_t)blockIdx.x * 128;
  int tid = threadIdx.x;

  const _Float16* src = in + m_base * 64;      // 16KB = 1024 x 16B
  unsigned lx = lds_addr_of(lX), lw = lds_addr_of(lW);
  for (int ck = tid; ck < 1024; ck += 256) async_copy16(lx + ck * 16, ck * 16, src);
  for (int ck = tid; ck < 512;  ck += 256) async_copy16(lw + ck * 16, ck * 16, wpk);
  wait_async();
  __syncthreads();

  int wv = tid >> 5, lane = tid & 31;
  int mL = lane & 15, hiL = lane >> 4;
  int m0 = wv * 16;

  v8f acc[4] = {};
#pragma unroll
  for (int c0 = 0; c0 < 64; c0 += 32) {
    v16h af = make_afrag(&lX[(size_t)(m0 + mL) * 64 + c0], hiL);
#pragma unroll
    for (int nt = 0; nt < 4; ++nt) {
      v16h bf = *(const v16h*)&lW[(size_t)(nt * 16 + mL) * 64 + c0 + 16 * hiL];
      acc[nt] = wmma_f16(af, bf, acc[nt]);
    }
  }
#pragma unroll
  for (int nt = 0; nt < 4; ++nt) {
    int n = nt * 16 + mL;
    float bv = bias[n];
#pragma unroll
    for (int r = 0; r < 8; ++r)
      out[(m_base + m0 + r + 8 * hiL) * 64 + n] = (_Float16)(acc[nt][r] + bv);
  }
}

// Same, but X is f32 (fold accumulator) -> convert while staging.
__global__ __launch_bounds__(256) void linear64_f2h(
    const float* __restrict__ in, const _Float16* __restrict__ wpk,
    const float* __restrict__ bias, _Float16* __restrict__ out) {
  __shared__ alignas(32) _Float16 lX[128 * 64];
  __shared__ alignas(32) _Float16 lW[64 * 64];
  size_t m_base = (size_t)blockIdx.x * 128;
  int tid = threadIdx.x;

  for (int idx = tid; idx < 128 * 64; idx += 256)
    lX[idx] = (_Float16)in[m_base * 64 + idx];
  unsigned lw = lds_addr_of(lW);
  for (int ck = tid; ck < 512; ck += 256) async_copy16(lw + ck * 16, ck * 16, wpk);
  wait_async();
  __syncthreads();

  int wv = tid >> 5, lane = tid & 31;
  int mL = lane & 15, hiL = lane >> 4;
  int m0 = wv * 16;

  v8f acc[4] = {};
#pragma unroll
  for (int c0 = 0; c0 < 64; c0 += 32) {
    v16h af = make_afrag(&lX[(size_t)(m0 + mL) * 64 + c0], hiL);
#pragma unroll
    for (int nt = 0; nt < 4; ++nt) {
      v16h bf = *(const v16h*)&lW[(size_t)(nt * 16 + mL) * 64 + c0 + 16 * hiL];
      acc[nt] = wmma_f16(af, bf, acc[nt]);
    }
  }
#pragma unroll
  for (int nt = 0; nt < 4; ++nt) {
    int n = nt * 16 + mL;
    float bv = bias[n];
#pragma unroll
    for (int r = 0; r < 8; ++r)
      out[(m_base + m0 + r + 8 * hiL) * 64 + n] = (_Float16)(acc[nt][r] + bv);
  }
}

// ---------------------------------------------------------------------------
// Fused attn: logits WMMA (N=648, B frags straight from packed-f16 global)
// -> softmax over 9 -> apply -> fold scatter (atomicAdd, f32 accumulator).
// One block = 16 pixels of one row.
// ---------------------------------------------------------------------------
__global__ __launch_bounds__(256) void attn_fused(
    const _Float16* __restrict__ x,     // BHWC f16 (features)
    const _Float16* __restrict__ v,     // BHWC f16 (values)
    const _Float16* __restrict__ wah,   // packed (648,64) f16
    const float* __restrict__ b_attn,   // (648) f32
    float* __restrict__ xw_fold) {      // BHWC f32, pre-zeroed
  __shared__ alignas(32) _Float16 lX[16 * 64];
  __shared__ float logits[16 * 656];

  int blk = blockIdx.x;
  int pgrp = blk & 7;
  int bh = blk >> 3;
  int b = bh >> 7, h = bh & 127;
  int p0 = pgrp * 16;
  int tid = threadIdx.x;
  size_t rowbase = ((size_t)b * 128 + h) * 128;

  {  // 16 pixels * 64 ch f16 = 2KB = 128 x 16B async
    const _Float16* src = x + (rowbase + p0) * 64;
    unsigned lx = lds_addr_of(lX);
    for (int ck = tid; ck < 128; ck += 256) async_copy16(lx + ck * 16, ck * 16, src);
  }
  wait_async();
  __syncthreads();

  int wv = tid >> 5, lane = tid & 31;
  int mL = lane & 15, hiL = lane >> 4;

  v16h afr[2];
#pragma unroll
  for (int kc = 0; kc < 2; ++kc)
    afr[kc] = make_afrag(&lX[(size_t)mL * 64 + kc * 32], hiL);

  for (int nt = wv; nt < 41; nt += 8) {
    int n0 = nt * 16;
    v8f acc = {};
#pragma unroll
    for (int kc = 0; kc < 2; ++kc) {
      int nrow = n0 + mL; if (nrow >= 648) nrow = 0;
      v16h bf = *(const v16h*)(wah + (size_t)nrow * 64 + kc * 32 + 16 * hiL);
      acc = wmma_f16(afr[kc], bf, acc);
    }
    int n = n0 + mL;
    if (n < 648) {
      float bv = b_attn[n];
#pragma unroll
      for (int r = 0; r < 8; ++r)
        logits[(r + 8 * hiL) * 656 + n] = acc[r] + bv;
    }
  }
  __syncthreads();

  const float scale = 0.35355339059327373f;  // hd^-0.5
  for (int row = tid; row < 1152; row += 256) {
    int px = row / 72, sub = row % 72;
    float* base = &logits[px * 656 + sub * 9];
    float t[9], mx = -1e30f;
#pragma unroll
    for (int q = 0; q < 9; ++q) { t[q] = base[q] * scale; mx = fmaxf(mx, t[q]); }
    float s = 0.f;
#pragma unroll
    for (int q = 0; q < 9; ++q) { t[q] = __expf(t[q] - mx); s += t[q]; }
    float rs = 1.f / s;
#pragma unroll
    for (int q = 0; q < 9; ++q) base[q] = t[q] * rs;
  }
  __syncthreads();

  int dh   = tid & 1;
  int head = (tid >> 1) & 7;
  int px   = tid >> 4;
  int wpix = p0 + px;
  int cbase = head * 8 + dh * 4;
  const float* probs = &logits[px * 656 + head * 81];

  float xw[9][4];
#pragma unroll
  for (int p = 0; p < 9; ++p)
#pragma unroll
    for (int d = 0; d < 4; ++d) xw[p][d] = 0.f;

  for (int q = 0; q < 9; ++q) {
    int nh = h + q / 3 - 1, nw = wpix + q % 3 - 1;
    float vq[4] = {0.f, 0.f, 0.f, 0.f};
    if (nh >= 0 && nh < 128 && nw >= 0 && nw < 128) {
      const _Float16* vp = &v[(((size_t)b * 128 + nh) * 128 + nw) * 64 + cbase];
      vq[0] = (float)vp[0]; vq[1] = (float)vp[1];
      vq[2] = (float)vp[2]; vq[3] = (float)vp[3];
    }
#pragma unroll
    for (int p = 0; p < 9; ++p) {
      float a = probs[p * 9 + q];
#pragma unroll
      for (int d = 0; d < 4; ++d) xw[p][d] += a * vq[d];
    }
  }
  for (int p = 0; p < 9; ++p) {
    int th = h + p / 3 - 1, tw = wpix + p % 3 - 1;
    if (th >= 0 && th < 128 && tw >= 0 && tw < 128) {
      float* op = &xw_fold[(((size_t)b * 128 + th) * 128 + tw) * 64 + cbase];
      for (int d = 0; d < 4; ++d) atomicAdd(op + d, xw[p][d]);
    }
  }
}

// ---------------------------------------------------------------------------
extern "C" void kernel_launch(void* const* d_in, const int* in_sizes, int n_in,
                              void* d_out, int out_size, void* d_ws, size_t ws_size,
                              hipStream_t stream) {
  const float* x      = (const float*)d_in[0];
  const float* conv_w = (const float*)d_in[1];
  const float* bn_g   = (const float*)d_in[2];
  const float* bn_b   = (const float*)d_in[3];
  const float* bn_m   = (const float*)d_in[4];
  const float* bn_v   = (const float*)d_in[5];
  const float* w_v    = (const float*)d_in[6];
  const float* b_v    = (const float*)d_in[7];
  const float* w_attn = (const float*)d_in[8];
  const float* b_attn = (const float*)d_in[9];
  const float* w_proj = (const float*)d_in[10];
  const float* b_proj = (const float*)d_in[11];
  float* out = (float*)d_out;

  const size_t ACT = 4ull * 128 * 128 * 64;
  float*     foldf = (float*)d_ws;                 // ACT f32 (fold accumulator)
  _Float16*  bufA  = (_Float16*)(foldf + ACT);     // ACT f16
  _Float16*  bufB  = bufA + ACT;                   // ACT f16
  _Float16*  bufC  = bufB + ACT;                   // ACT f16
  _Float16*  wpk   = bufC + ACT;                   // 197120 halves packed weights
  _Float16*  wlin  = wpk + 147456;                 // w_v | w_proj
  _Float16*  wah   = wpk + 155648;                 // w_attn
  dim3 blk(256);

  pack_weights<<<770, blk, 0, stream>>>(conv_w, w_v, w_proj, w_attn, wpk);
  nchw_to_bhwc_h<<<16384, blk, 0, stream>>>(x, bufA);

  conv3_bn_relu<<<512, blk, 0, stream>>>(bufA, wpk + 0 * 36864,
      bn_g + 0,   bn_b + 0,   bn_m + 0,   bn_v + 0,   (void*)bufB, 0);
  conv3_bn_relu<<<512, blk, 0, stream>>>(bufB, wpk + 1 * 36864,
      bn_g + 64,  bn_b + 64,  bn_m + 64,  bn_v + 64,  (void*)bufC, 0);

  linear64_h<<<512, blk, 0, stream>>>(bufC, wlin, b_v, bufA);         // v

  hipMemsetAsync(foldf, 0, ACT * sizeof(float), stream);
  attn_fused<<<4096, blk, 0, stream>>>(bufC, bufA, wah, b_attn, foldf);

  linear64_f2h<<<512, blk, 0, stream>>>(foldf, wlin + 4096, b_proj, bufB);

  conv3_bn_relu<<<512, blk, 0, stream>>>(bufB, wpk + 2 * 36864,
      bn_g + 128, bn_b + 128, bn_m + 128, bn_v + 128, (void*)bufA, 0);
  conv3_bn_relu<<<512, blk, 0, stream>>>(bufA, wpk + 3 * 36864,
      bn_g + 192, bn_b + 192, bn_m + 192, bn_v + 192, (void*)out, 1);  // NCHW f32
}
